// SparseSelfAttention_10230612099477
// MI455X (gfx1250) — matrix operand order repaired and checked by reading, not compiled, not run on hardware
//
#include <hip/hip_runtime.h>
#include <hip/hip_bf16.h>

typedef __bf16 bf16;
typedef __attribute__((ext_vector_type(16))) __bf16 v16bf;
typedef __attribute__((ext_vector_type(8)))  __bf16 v8bf;
typedef __attribute__((ext_vector_type(8)))  float  v8f;
typedef __attribute__((ext_vector_type(4)))  unsigned int v4u;
typedef __attribute__((ext_vector_type(8)))  int v8i;
typedef __attribute__((ext_vector_type(4)))  int v4i;

namespace {
constexpr int    kDk     = 64;     // key size == head size
constexpr int    kSeq    = 4096;
constexpr int    kD      = 512;
constexpr int    kN      = 1025;   // dilated sequence length (incl. zero pad row)
constexpr int    kNP     = 1056;   // padded to 33*32 (WMMA K blocks)
constexpr int    kGroups = 64;     // batch * heads * rate
constexpr size_t kGSZ    = (size_t)kGroups * kNP * kDk;  // elems per q/k/v buffer
}

__device__ __forceinline__ v8f wmma_bf16(v16bf a, v16bf b, v8f c) {
  return __builtin_amdgcn_wmma_f32_16x16x32_bf16(false, a, false, b, (short)0, c,
                                                 false, false);
}

// 16 contiguous bf16 -> v16bf  (contiguous k order: lane holds K=e or K=16+e)
__device__ __forceinline__ v16bf load16c(const bf16* p) {
  v8bf a = *(const v8bf*)p;
  v8bf b = *(const v8bf*)(p + 8);
  return __builtin_shufflevector(a, b, 0, 1, 2, 3, 4, 5, 6, 7,
                                       8, 9, 10, 11, 12, 13, 14, 15);
}
// A-operand pattern: elems 0..7 = p[0..7], elems 8..15 = p[16..23]
__device__ __forceinline__ v16bf load8x2(const bf16* p) {
  v8bf a = *(const v8bf*)p;
  v8bf b = *(const v8bf*)(p + 16);
  return __builtin_shufflevector(a, b, 0, 1, 2, 3, 4, 5, 6, 7,
                                       8, 9, 10, 11, 12, 13, 14, 15);
}

// TDM: DMA one 16x64 bf16 tile (row-major, stride 64) from global into LDS.
// Descriptor packing per CDNA5 ISA.
// This toolchain (clang-23 / therock HIP headers) uses the 6-arg builtin:
//   (uint32x4 g0, int32x8 g1, int32x4, int32x4, int32x8, i32 cpol)
__device__ __forceinline__ void tdm_load_ktile(const bf16* gsrc, bf16* lds_dst) {
  unsigned long long ga = (unsigned long long)(uintptr_t)gsrc;
  unsigned la = (unsigned)(uintptr_t)lds_dst;  // low 32 bits = LDS byte offset
  v4u g0;
  g0.x = 1u;                                           // count=1 (valid user D#)
  g0.y = la;                                           // lds_addr
  g0.z = (unsigned)(ga & 0xffffffffu);                 // global_addr[31:0]
  g0.w = (unsigned)((ga >> 32) & 0x01ffffffu) | (2u << 30);  // addr[56:32], type=2
  v8i g1;
  g1[0] = (int)(1u << 16);   // workgroup_mask=0, data_size=1 (2 bytes)
  g1[1] = (int)(64u << 16);  // tensor_dim0 = 64  (bits 79:48, low part)
  g1[2] = (int)(16u << 16);  // tensor_dim1 = 16  (bits 111:80, low part)
  g1[3] = (int)(64u << 16);  // tile_dim0 = 64    (bits 127:112)
  g1[4] = 16;                // tile_dim1 = 16    (bits 143:128); tile_dim2 = 0
  g1[5] = 64;                // tensor_dim0_stride = 64 (bits 207:160, low 32)
  g1[6] = 0;
  g1[7] = 0;
  v4i z4 = {0, 0, 0, 0};               // 2D tensor: groups 2/3 unused
  v8i z8 = {0, 0, 0, 0, 0, 0, 0, 0};
  __builtin_amdgcn_tensor_load_to_lds(g0, g1, z4, z4, z8, 0);
}

// ---------------------------------------------------------------------------
// Kernel 1: transpose weights to bf16, Wt[mat][out][in]
// ---------------------------------------------------------------------------
__global__ void prep_weights(const float* __restrict__ Wq,
                             const float* __restrict__ Wk,
                             const float* __restrict__ Wv,
                             bf16* __restrict__ Wt) {
  int idx = blockIdx.x * blockDim.x + threadIdx.x;
  if (idx >= 3 * kD * kD) return;
  int mat = idx / (kD * kD);
  int rem = idx % (kD * kD);
  int o   = rem / kD;
  int i   = rem % kD;
  const float* W = (mat == 0) ? Wq : (mat == 1) ? Wk : Wv;
  Wt[idx] = (bf16)W[(size_t)i * kD + o];
}

// ---------------------------------------------------------------------------
// Kernel 2: projections. Writes qg/kg [g][i][64] and vt [g][64][i] (bf16),
// already split into dilation groups; zero for padded positions.
// grid = (17 itile-quads, 64 groups, 3 mats), block = 128 (4 waves, 1 itile/wave)
// ---------------------------------------------------------------------------
__global__ void proj_kernel(const float* __restrict__ x,
                            const bf16* __restrict__ Wt,
                            bf16* __restrict__ qg,
                            bf16* __restrict__ kg,
                            bf16* __restrict__ vt) {
  const int wid   = threadIdx.x >> 5;
  const int itile = blockIdx.x * 4 + wid;
  if (itile >= 66) return;  // wave-uniform exit
  const int g    = blockIdx.y;
  const int mat  = blockIdx.z;
  const int lane = threadIdx.x & 31;
  const int rlo  = lane & 15;
  const bool hi  = lane >= 16;
  const int bb = g >> 5, h = (g >> 2) & 7, rr = g & 3;
  const int i0 = itile * 16;

  // A-operand source row for this lane (matrix row = rlo)
  const int  tA     = 4 * (i0 + rlo) + rr;
  const bool aValid = (tA < kSeq);
  const float* xrow = x + ((size_t)bb * kSeq + (aValid ? tA : 0)) * kD;

  const bf16* wbase = Wt + ((size_t)mat * kD + h * kDk) * kD;

  v8f acc[4];
#pragma unroll
  for (int c = 0; c < 4; ++c)
    acc[c] = v8f{0.f, 0.f, 0.f, 0.f, 0.f, 0.f, 0.f, 0.f};

  for (int kc = 0; kc < 16; ++kc) {
    const int k0 = kc * 32;
    const int ka = k0 + (hi ? 8 : 0);
    v16bf a;
    if (aValid) {
#pragma unroll
      for (int e = 0; e < 8; ++e) {
        a[e]     = (bf16)xrow[ka + e];
        a[e + 8] = (bf16)xrow[ka + 16 + e];
      }
    } else {
#pragma unroll
      for (int e = 0; e < 16; ++e) a[e] = (bf16)0.f;
    }
#pragma unroll
    for (int c = 0; c < 4; ++c) {
      const bf16* wrow = wbase + (size_t)(c * 16 + rlo) * kD + k0 + (hi ? 8 : 0);
      v16bf b = load8x2(wrow);
      acc[c]  = wmma_bf16(a, b, acc[c]);
    }
  }

#pragma unroll
  for (int c = 0; c < 4; ++c) {
    if (mat == 2) {
      // vt[g][col][i]: contiguous in i -> pack 8 bf16 and store 16B
      v8bf vs;
#pragma unroll
      for (int r = 0; r < 8; ++r) vs[r] = (bf16)acc[c][r];
      *(v8bf*)(vt + ((size_t)g * kDk + c * 16 + rlo) * kNP + i0 + (hi ? 8 : 0)) = vs;
    } else {
      bf16* dst = (mat == 0 ? qg : kg);
#pragma unroll
      for (int r = 0; r < 8; ++r) {
        const int m = r + (hi ? 8 : 0);
        dst[((size_t)g * kNP + i0 + m) * kDk + c * 16 + rlo] = (bf16)acc[c][r];
      }
    }
  }
}

// ---------------------------------------------------------------------------
// Kernel 3: attention per (group, 16-query tile).
// grid = (64, 64), block = 128 (4 waves cooperating on one score buffer).
// Dynamic LDS: 16 * kNP floats. Static LDS: per-wave double-buffered K tiles
// staged via the Tensor Data Mover.
// ---------------------------------------------------------------------------
__global__ void attn_kernel(const bf16* __restrict__ qg,
                            const bf16* __restrict__ kg,
                            const bf16* __restrict__ vt,
                            float* __restrict__ out) {
  extern __shared__ float sc[];                 // [16][kNP] score rows
  __shared__ bf16 kstage[4][2][16 * kDk];       // per-wave double buffer (16 KB)

  const int qt   = blockIdx.x;
  const int g    = blockIdx.y;
  const int wid  = threadIdx.x >> 5;
  const int lane = threadIdx.x & 31;
  const int rlo  = lane & 15;
  const bool hi  = lane >= 16;
  const int q0   = qt * 16;

  // Q as A-operand (dk=64 -> two 32-wide K chunks); same tile for all 4 waves
  const bf16* qrow = qg + ((size_t)g * kNP + q0 + rlo) * kDk;
  const v16bf aq0 = load8x2(qrow + (hi ? 8 : 0));
  const v16bf aq1 = load8x2(qrow + 32 + (hi ? 8 : 0));

  // ---- pass 1: S = Q K^T / 8. Wave w handles key tiles w, w+4, ... ----
  // K tiles are DMA'd global->LDS by the TDM, double buffered per wave.
  const bf16* kgrp = kg + (size_t)g * kNP * kDk;
  int ib = 0;
  if (wid < 66) tdm_load_ktile(kgrp + (size_t)wid * 16 * kDk, &kstage[wid][0][0]);
  for (int kt = wid; kt < 66; kt += 4) {
    const int ktn = kt + 4;
    if (ktn < 66)
      tdm_load_ktile(kgrp + (size_t)ktn * 16 * kDk, &kstage[wid][ib ^ 1][0]);
    if (ktn < 66)
      __builtin_amdgcn_s_wait_tensorcnt(1);  // oldest (current) tile landed
    else
      __builtin_amdgcn_s_wait_tensorcnt(0);

    const bf16* kr = &kstage[wid][ib][rlo * kDk];
    v16bf b0 = load8x2(kr + (hi ? 8 : 0));
    v16bf b1 = load8x2(kr + 32 + (hi ? 8 : 0));
    v8f s = v8f{0.f, 0.f, 0.f, 0.f, 0.f, 0.f, 0.f, 0.f};
    s = wmma_bf16(aq0, b0, s);
    s = wmma_bf16(aq1, b1, s);
    const int key = kt * 16 + rlo;
#pragma unroll
    for (int r = 0; r < 8; ++r) {
      const int m = r + (hi ? 8 : 0);
      sc[(size_t)m * kNP + key] = (key < kN) ? s[r] * 0.125f : -1e30f;
    }
    ib ^= 1;
  }
  __syncthreads();

  // ---- pass 2: softmax over [n main + 7 band duplicates], fold band back.
  //      Wave w owns rows w, w+4, w+8, w+12. ----
  for (int m = wid; m < 16; m += 4) {
    const int qi = q0 + m;
    float* srow = sc + (size_t)m * kNP;
    float band[7];
#pragma unroll
    for (int j = 0; j < 7; ++j) {
      const int mi = qi + j - 3;
      band[j] = (mi >= 0 && mi < kN) ? srow[mi] : 0.0f;  // zero-padded patch
    }
    float mx = -1e30f;
    for (int k = lane; k < kNP; k += 32) mx = fmaxf(mx, srow[k]);
#pragma unroll
    for (int j = 0; j < 7; ++j) mx = fmaxf(mx, band[j]);
#pragma unroll
    for (int off = 16; off > 0; off >>= 1) mx = fmaxf(mx, __shfl_xor(mx, off, 32));

    float sum = 0.f;
    for (int k = lane; k < kNP; k += 32) {
      const float e = __expf(srow[k] - mx);
      srow[k] = e;
      sum += e;
    }
    if (lane == 0) {
#pragma unroll
      for (int j = 0; j < 7; ++j) sum += __expf(band[j] - mx);
    }
#pragma unroll
    for (int off = 16; off > 0; off >>= 1) sum += __shfl_xor(sum, off, 32);
    const float inv = 1.0f / sum;
    for (int k = lane; k < kNP; k += 32) srow[k] *= inv;
    // fold duplicate band weights back into the owning row (same wave: LDS in-order)
    if (lane < 7) {
      const int mi = qi + lane - 3;
      if (mi >= 0 && mi < kN) {
        float bv = 0.f;
#pragma unroll
        for (int j = 0; j < 7; ++j)
          if (lane == j) bv = band[j];
        srow[mi] += __expf(bv - mx) * inv;
      }
    }
  }
  __syncthreads();

  // ---- pass 3: O = P V. Wave w computes dv chunk c = w over all keys. ----
  v8f o = v8f{0.f, 0.f, 0.f, 0.f, 0.f, 0.f, 0.f, 0.f};
  const bf16* vchunk = vt + ((size_t)g * kDk + wid * 16 + rlo) * kNP;
  for (int kp2 = 0; kp2 < 33; ++kp2) {
    const int kb = kp2 * 32;
    const float* pr = sc + (size_t)rlo * kNP + kb + (hi ? 8 : 0);
    v16bf ap;
#pragma unroll
    for (int e = 0; e < 8; ++e) {
      ap[e]     = (bf16)pr[e];
      ap[e + 8] = (bf16)pr[16 + e];
    }
    const bf16* vr = vchunk + kb + (hi ? 8 : 0);
    if (kp2 + 1 < 33) __builtin_prefetch(vr + 32, 0, 0);
    o = wmma_bf16(ap, load8x2(vr), o);
  }

  // ---- store to out[b][t][h*64 + col], t = 4*i + rr (always < 4096 here) ----
  const int bb = g >> 5, h = (g >> 2) & 7, rr = g & 3;
#pragma unroll
  for (int r = 0; r < 8; ++r) {
    const int m = r + (hi ? 8 : 0);
    const int t = 4 * (q0 + m) + rr;
    out[((size_t)bb * kSeq + t) * kD + h * kDk + wid * 16 + rlo] = o[r];
  }
}

// ---------------------------------------------------------------------------
extern "C" void kernel_launch(void* const* d_in, const int* in_sizes, int n_in,
                              void* d_out, int out_size, void* d_ws, size_t ws_size,
                              hipStream_t stream) {
  (void)in_sizes; (void)n_in; (void)out_size; (void)ws_size;
  const float* x  = (const float*)d_in[0];
  const float* Wq = (const float*)d_in[1];
  const float* Wk = (const float*)d_in[2];
  const float* Wv = (const float*)d_in[3];
  float* out = (float*)d_out;

  bf16* qg = (bf16*)d_ws;
  bf16* kg = qg + kGSZ;
  bf16* vt = kg + kGSZ;
  bf16* Wt = vt + kGSZ;

  prep_weights<<<(3 * kD * kD + 255) / 256, 256, 0, stream>>>(Wq, Wk, Wv, Wt);
  proj_kernel<<<dim3(17, kGroups, 3), 128, 0, stream>>>(x, Wt, qg, kg, vt);
  attn_kernel<<<dim3(64, kGroups), 128, (size_t)16 * kNP * sizeof(float), stream>>>(
      qg, kg, vt, out);
}
